// MultiheadedAttention_39994735460574
// MI455X (gfx1250) — matrix operand order repaired and checked by reading, not compiled, not run on hardware
//
#include <hip/hip_runtime.h>
#include <hip/hip_bf16.h>

typedef __attribute__((ext_vector_type(16))) _Float16 v16h;
typedef __attribute__((ext_vector_type(8)))  float    v8f;

static constexpr int Bc = 2, CIN = 512, Lc = 1024, Hc = 32, Dc = 64, Ac = 2048;

// ---------- CDNA5 async global->LDS copy via inline asm (ISA 08_async_tensor) ----------
#if defined(__HIP_DEVICE_COMPILE__) && defined(__gfx1250__)
#define USE_ASYNC_ASM 1
#else
#define USE_ASYNC_ASM 0
#endif

// copy 16 bytes (8 halves) global -> LDS, tracked by ASYNCcnt
__device__ __forceinline__ void cp16(_Float16* dst, const _Float16* src) {
#if USE_ASYNC_ASM
  unsigned ldsoff = (unsigned)(unsigned long long)(const void*)dst;
  asm volatile("global_load_async_to_lds_b128 %0, %1, off"
               :
               : "v"(ldsoff), "v"((unsigned long long)(const void*)src)
               : "memory");
#else
  *(uint4*)dst = *(const uint4*)src;
#endif
}

__device__ __forceinline__ void async_wait_all() {
#if USE_ASYNC_ASM
  asm volatile("s_wait_asynccnt 0" ::: "memory");
#endif
}

// ---- WMMA fragment helpers (CDNA5 16x16x32 f16, wave32 layouts per ISA 7.12.2) ----

__device__ __forceinline__ v16h load_a_frag(const _Float16* lds, int ldm, int mBase, int kBase) {
  const int lane = threadIdx.x & 31;
  const int half = lane >> 4;
  const _Float16* p = lds + (size_t)(mBase + (lane & 15)) * ldm + kBase + half * 8;
  v16h a;
#pragma unroll
  for (int i = 0; i < 8; ++i) a[i] = p[i];
#pragma unroll
  for (int i = 0; i < 8; ++i) a[8 + i] = p[16 + i];
  return a;
}

__device__ __forceinline__ v16h load_b_frag(const _Float16* ldsBT, int ldk, int nBase, int kBase) {
  const int lane = threadIdx.x & 31;
  const int half = lane >> 4;
  const _Float16* p = ldsBT + (size_t)(nBase + (lane & 15)) * ldk + kBase + half * 8;
  v16h b;
#pragma unroll
  for (int i = 0; i < 8; ++i) b[i] = p[i];
#pragma unroll
  for (int i = 0; i < 8; ++i) b[8 + i] = p[16 + i];
  return b;
}

__device__ __forceinline__ v8f wmma_f16(v16h a, v16h b, v8f c) {
  return __builtin_amdgcn_wmma_f32_16x16x32_f16(false, a, false, b, (short)0, c, false, false);
}

// stage x^T tile [rows l x 32 c] (f32 strided -> f16 LDS), float4 loads, coalesced on l
template <int ROWS>
__device__ __forceinline__ void stage_xT(_Float16* dstT, const float* x, size_t rowBase,
                                         int kb, int lBase, int tid) {
  constexpr int CH = ROWS / 4 * 32;  // float4 chunks
#pragma unroll
  for (int it = 0; it < CH / 256; ++it) {
    int e4 = tid + it * 256;
    int lgrp = e4 % (ROWS / 4), c = e4 / (ROWS / 4);
    const float4 vv = *(const float4*)&x[(rowBase + kb + c) * Lc + lBase + lgrp * 4];
    _Float16* q = &dstT[(lgrp * 4) * 32 + c];
    q[0] = (_Float16)vv.x;
    q[32] = (_Float16)vv.y;
    q[64] = (_Float16)vv.z;
    q[96] = (_Float16)vv.w;
  }
}

// stage row-major f32 weight tile [ROWS x 32] -> f16 LDS, float4 loads
template <int ROWS>
__device__ __forceinline__ void stage_w(_Float16* dst, const float* W, size_t rowBase,
                                        size_t rowStride, int kb, int tid) {
  constexpr int CH = ROWS * 8;
#pragma unroll
  for (int it = 0; it < CH / 256; ++it) {
    int e4 = tid + it * 256;
    int cg = e4 & 7, rr = e4 >> 3;
    const float4 vv = *(const float4*)&W[(rowBase + rr) * rowStride + kb + cg * 4];
    _Float16* q = &dst[rr * 32 + cg * 4];
    q[0] = (_Float16)vv.x;
    q[1] = (_Float16)vv.y;
    q[2] = (_Float16)vv.z;
    q[3] = (_Float16)vv.w;
  }
}

// ================== Kernel 1: Q/K projections ==================
__global__ __launch_bounds__(256) void proj_qk_kernel(
    const float* __restrict__ x, const float* __restrict__ Wq, const float* __restrict__ bq,
    const float* __restrict__ Wk, const float* __restrict__ bk,
    _Float16* __restrict__ q_ws, _Float16* __restrict__ k_ws) {
  __shared__ __align__(16) _Float16 At[128 * 32];  // x^T tile [l x c]
  __shared__ __align__(16) _Float16 Bt[64 * 32];   // W tile [d x c]
  const int tid = threadIdx.x;
  const int lBase = blockIdx.x * 128;
  const int proj = blockIdx.y >> 6;
  const int bh = blockIdx.y & 63;
  const int b = bh >> 5, h = bh & 31;
  const float* W = proj ? Wk : Wq;
  const float* bias = proj ? bk : bq;
  _Float16* out = proj ? k_ws : q_ws;
  const int wave = tid >> 5, lane = tid & 31;
  const int wr = wave >> 1, wc = wave & 1;  // wave owns 32x32 of 128x64 output
  v8f acc[2][2] = {};
  for (int kb = 0; kb < CIN; kb += 32) {
    __syncthreads();
    stage_xT<128>(At, x, (size_t)b * CIN, kb, lBase, tid);
    stage_w<64>(Bt, W, (size_t)h * Dc, CIN, kb, tid);
    if (kb + 32 < CIN) {
      __builtin_prefetch(&x[((size_t)b * CIN + kb + 32 + (tid >> 5)) * Lc + lBase + (tid & 31) * 4], 0, 3);
      __builtin_prefetch(&W[((size_t)h * Dc + (tid >> 2)) * CIN + kb + 32 + (tid & 3) * 8], 0, 3);
    }
    __syncthreads();
    v16h a0 = load_a_frag(At, 32, 32 * wr, 0);
    v16h a1 = load_a_frag(At, 32, 32 * wr + 16, 0);
    v16h b0 = load_b_frag(Bt, 32, 32 * wc, 0);
    v16h b1 = load_b_frag(Bt, 32, 32 * wc + 16, 0);
    acc[0][0] = wmma_f16(a0, b0, acc[0][0]);
    acc[0][1] = wmma_f16(a0, b1, acc[0][1]);
    acc[1][0] = wmma_f16(a1, b0, acc[1][0]);
    acc[1][1] = wmma_f16(a1, b1, acc[1][1]);
  }
#pragma unroll
  for (int i = 0; i < 2; ++i)
#pragma unroll
    for (int j = 0; j < 2; ++j)
#pragma unroll
      for (int r = 0; r < 8; ++r) {
        int m = 32 * wr + 16 * i + (lane >> 4) * 8 + r;  // local l
        int n = 32 * wc + 16 * j + (lane & 15);          // d
        float v = acc[i][j][r] + bias[h * Dc + n];
        out[((size_t)bh * Lc + lBase + m) * Dc + n] = (_Float16)v;
      }
}

// ================== Kernel 2: V projection ==================
__global__ __launch_bounds__(256) void proj_v_kernel(
    const float* __restrict__ x, const float* __restrict__ Wv, const float* __restrict__ bv,
    _Float16* __restrict__ v_ws) {
  __shared__ __align__(16) _Float16 At[64 * 32];   // Wv tile [d x c]
  __shared__ __align__(16) _Float16 Bt[128 * 32];  // x^T tile [l x c]
  const int tid = threadIdx.x;
  const int lBase = blockIdx.x * 128;
  const int bh = blockIdx.y;
  const int b = bh >> 5, h = bh & 31;
  const int wave = tid >> 5, lane = tid & 31;
  const int wr = wave >> 2, wc = wave & 3;  // 64x128 out, wave owns 32x32
  v8f acc[2][2] = {};
  for (int kb = 0; kb < CIN; kb += 32) {
    __syncthreads();
    stage_w<64>(At, Wv, (size_t)h * Dc, CIN, kb, tid);
    stage_xT<128>(Bt, x, (size_t)b * CIN, kb, lBase, tid);
    if (kb + 32 < CIN) {
      __builtin_prefetch(&x[((size_t)b * CIN + kb + 32 + (tid >> 5)) * Lc + lBase + (tid & 31) * 4], 0, 3);
      __builtin_prefetch(&Wv[((size_t)h * Dc + (tid >> 2)) * CIN + kb + 32 + (tid & 3) * 8], 0, 3);
    }
    __syncthreads();
    v16h a0 = load_a_frag(At, 32, 32 * wr, 0);
    v16h a1 = load_a_frag(At, 32, 32 * wr + 16, 0);
    v16h b0 = load_b_frag(Bt, 32, 32 * wc, 0);
    v16h b1 = load_b_frag(Bt, 32, 32 * wc + 16, 0);
    acc[0][0] = wmma_f16(a0, b0, acc[0][0]);
    acc[0][1] = wmma_f16(a0, b1, acc[0][1]);
    acc[1][0] = wmma_f16(a1, b0, acc[1][0]);
    acc[1][1] = wmma_f16(a1, b1, acc[1][1]);
  }
#pragma unroll
  for (int i = 0; i < 2; ++i)
#pragma unroll
    for (int j = 0; j < 2; ++j)
#pragma unroll
      for (int r = 0; r < 8; ++r) {
        int d = 32 * wr + 16 * i + (lane >> 4) * 8 + r;
        int l = lBase + 32 * wc + 16 * j + (lane & 15);
        float v = acc[i][j][r] + bv[h * Dc + d];
        v_ws[((size_t)bh * Dc + d) * Lc + l] = (_Float16)v;
      }
}

// ================== Kernel 3: softmax row stats (pre-mask, with s_rel) ==================
__global__ __launch_bounds__(256) void attn_stats_kernel(
    const _Float16* __restrict__ q_ws, const _Float16* __restrict__ k_ws,
    float* __restrict__ rmax_g, float* __restrict__ rsum_g) {
  __shared__ __align__(16) _Float16 qT[128 * 64];
  __shared__ __align__(16) _Float16 kT[64 * 64];
  __shared__ float S[128 * 64];
  __shared__ float rowmax[128], rowsum[128], tmp[256];
  const int tid = threadIdx.x;
  const int bh = blockIdx.y;
  const int lBase = blockIdx.x * 128;
  {
    const _Float16* qsrc = q_ws + ((size_t)bh * Lc + lBase) * Dc;
#pragma unroll
    for (int it = 0; it < 4; ++it) {
      int e8 = tid + it * 256;
      cp16(qT + e8 * 8, qsrc + e8 * 8);
    }
  }
  if (tid < 128) { rowmax[tid] = -1e30f; rowsum[tid] = 0.f; }
  const int wave = tid >> 5, lane = tid & 31;
  for (int mb = 0; mb < Lc; mb += 64) {
    __syncthreads();
    {
      const _Float16* ksrc = k_ws + ((size_t)bh * Lc + mb) * Dc;
#pragma unroll
      for (int it = 0; it < 2; ++it) {
        int e8 = tid + it * 256;
        cp16(kT + e8 * 8, ksrc + e8 * 8);
      }
    }
    async_wait_all();
    __syncthreads();
    v16h a0 = load_a_frag(qT, 64, 16 * wave, 0);
    v16h a1 = load_a_frag(qT, 64, 16 * wave, 32);
#pragma unroll
    for (int j = 0; j < 4; ++j) {
      v8f s = {};
      s = wmma_f16(a0, load_b_frag(kT, 64, 16 * j, 0), s);
      s = wmma_f16(a1, load_b_frag(kT, 64, 16 * j, 32), s);
#pragma unroll
      for (int r = 0; r < 8; ++r) {
        int lrow = 16 * wave + (lane >> 4) * 8 + r;
        int mcol = 16 * j + (lane & 15);
        int lg = lBase + lrow, mg = mb + mcol;
        S[lrow * 64 + mcol] = (s[r] - fabsf((float)(lg - mg))) * 0.125f;
      }
    }
    __syncthreads();
    {  // per-row online max/sum update (2 threads per row)
      int row = tid >> 1, base = (tid & 1) * 32;
      float mx = -1e30f;
#pragma unroll 8
      for (int c = 0; c < 32; ++c) mx = fmaxf(mx, S[row * 64 + base + c]);
      tmp[tid] = mx;
    }
    __syncthreads();
    if (tid < 128) {
      float nm = fmaxf(rowmax[tid], fmaxf(tmp[2 * tid], tmp[2 * tid + 1]));
      rowsum[tid] *= __expf(rowmax[tid] - nm);
      rowmax[tid] = nm;
    }
    __syncthreads();
    {
      int row = tid >> 1, base = (tid & 1) * 32;
      float sm = 0.f, rm = rowmax[row];
#pragma unroll 8
      for (int c = 0; c < 32; ++c) sm += __expf(S[row * 64 + base + c] - rm);
      tmp[tid] = sm;
    }
    __syncthreads();
    if (tid < 128) rowsum[tid] += tmp[2 * tid] + tmp[2 * tid + 1];
  }
  __syncthreads();
  if (tid < 128) {
    rmax_g[(size_t)bh * Lc + lBase + tid] = rowmax[tid];
    rsum_g[(size_t)bh * Lc + lBase + tid] = rowsum[tid];
  }
}

// ================== Kernel 4: fused normalize+mask+PV, ReLU, concat heads ==================
__global__ __launch_bounds__(256) void attn_out_kernel(
    const _Float16* __restrict__ q_ws, const _Float16* __restrict__ k_ws,
    const _Float16* __restrict__ v_ws,
    const float* __restrict__ rmax_g, const float* __restrict__ rsum_g,
    _Float16* __restrict__ h_ws) {
  __shared__ __align__(16) _Float16 kT[128 * 64];   // B^T rows m (K=d), fixed per block
  __shared__ __align__(16) _Float16 qT[64 * 64];    // A rows l chunk (K=d)
  __shared__ __align__(16) _Float16 vT[64 * 64];    // A rows d (K=l chunk)
  __shared__ __align__(16) _Float16 pBT[128 * 64];  // B^T rows m (K=l chunk), f16 probs
  __shared__ float rmaxc[64], rsumc[64];
  const int tid = threadIdx.x;
  const int bh = blockIdx.y;
  const int b = bh >> 5, h = bh & 31;
  const int mBase = blockIdx.x * 128;
  {
    const _Float16* ksrc = k_ws + ((size_t)bh * Lc + mBase) * Dc;
#pragma unroll
    for (int it = 0; it < 4; ++it) {
      int e8 = tid + it * 256;
      cp16(kT + e8 * 8, ksrc + e8 * 8);
    }
  }
  const int wave = tid >> 5, lane = tid & 31;
  const int sr = wave >> 1, sc = wave & 1;  // score tile 64x128: wave = 16 rows x 64 cols
  const int wr = wave >> 2, wc = wave & 3;  // out tile 64x128: wave = 32x32
  v8f oacc[2][2] = {};
  const int nChunks = 2 * blockIdx.x + 2;  // triu: only l-chunks with l <= mMax
  for (int lb = 0; lb < nChunks; ++lb) {
    const int lBase = lb * 64;
    __syncthreads();
    {
      const _Float16* qsrc = q_ws + ((size_t)bh * Lc + lBase) * Dc;
#pragma unroll
      for (int it = 0; it < 2; ++it) {
        int e8 = tid + it * 256;
        cp16(qT + e8 * 8, qsrc + e8 * 8);
        int d = (e8 * 8) >> 6, lo = (e8 * 8) & 63;
        cp16(vT + e8 * 8, v_ws + ((size_t)bh * Dc + d) * Lc + lBase + lo);
      }
    }
    if (tid < 64) {
      rmaxc[tid] = rmax_g[(size_t)bh * Lc + lBase + tid];
      rsumc[tid] = rsum_g[(size_t)bh * Lc + lBase + tid];
    }
    async_wait_all();
    __syncthreads();
    // scores [64 l x 128 m] -> normalized masked p stored transposed into pBT
    v16h a0 = load_a_frag(qT, 64, 16 * sr, 0);
    v16h a1 = load_a_frag(qT, 64, 16 * sr, 32);
#pragma unroll
    for (int j = 0; j < 4; ++j) {
      int nb = 64 * sc + 16 * j;
      v8f s = {};
      s = wmma_f16(a0, load_b_frag(kT, 64, nb, 0), s);
      s = wmma_f16(a1, load_b_frag(kT, 64, nb, 32), s);
#pragma unroll
      for (int r = 0; r < 8; ++r) {
        int lrow = 16 * sr + (lane >> 4) * 8 + r;
        int mcol = nb + (lane & 15);
        int lg = lBase + lrow, mg = mBase + mcol;
        float sco = (s[r] - fabsf((float)(lg - mg))) * 0.125f;
        float p = (lg <= mg) ? __expf(sco - rmaxc[lrow]) / rsumc[lrow] : 0.f;
        pBT[mcol * 64 + lrow] = (_Float16)p;
      }
    }
    __syncthreads();
    // o[d,m] += v[d, lchunk] @ p[lchunk, m]
#pragma unroll
    for (int ks = 0; ks < 64; ks += 32) {
      v16h av0 = load_a_frag(vT, 64, 32 * wr, ks);
      v16h av1 = load_a_frag(vT, 64, 32 * wr + 16, ks);
      v16h bp0 = load_b_frag(pBT, 64, 32 * wc, ks);
      v16h bp1 = load_b_frag(pBT, 64, 32 * wc + 16, ks);
      oacc[0][0] = wmma_f16(av0, bp0, oacc[0][0]);
      oacc[0][1] = wmma_f16(av0, bp1, oacc[0][1]);
      oacc[1][0] = wmma_f16(av1, bp0, oacc[1][0]);
      oacc[1][1] = wmma_f16(av1, bp1, oacc[1][1]);
    }
  }
#pragma unroll
  for (int i = 0; i < 2; ++i)
#pragma unroll
    for (int j = 0; j < 2; ++j)
#pragma unroll
      for (int r = 0; r < 8; ++r) {
        int d = 32 * wr + 16 * i + (lane >> 4) * 8 + r;
        int mg = mBase + 32 * wc + 16 * j + (lane & 15);
        float v = fmaxf(oacc[i][j][r], 0.f);  // per-head ReLU
        h_ws[((size_t)b * Lc + mg) * Ac + h * Dc + d] = (_Float16)v;  // [b][l][c] f16
      }
}

// ================== Kernel 5: final 1x1 convs (2048x2048 GEMM) ==================
__global__ __launch_bounds__(256) void final_gemm_kernel(
    const float* __restrict__ W, const float* __restrict__ bias,
    const _Float16* __restrict__ inT,  // [b][l][c] f16
    _Float16* __restrict__ outT,       // mode0: [b][l][o] f16 (ReLU)
    float* __restrict__ outF,          // mode1: [b][o][l] f32
    int mode) {
  __shared__ __align__(16) _Float16 At[128 * 32];
  __shared__ __align__(16) _Float16 Bt[128 * 32];
  const int tid = threadIdx.x;
  const int lBase = blockIdx.x * 128;
  const int oBase = blockIdx.y * 128;
  const int b = blockIdx.z;
  const int wave = tid >> 5, lane = tid & 31;
  const int wr = wave >> 1, wc = wave & 1;  // wave owns 32x64 of 128x128
  v8f acc[2][4] = {};
  for (int kb = 0; kb < Ac; kb += 32) {
    __syncthreads();
    stage_w<128>(At, W, (size_t)oBase, Ac, kb, tid);
#pragma unroll
    for (int it = 0; it < 2; ++it) {  // 128 rows x 4 chunks of 8 halves
      int e8 = tid + it * 256;
      int l = e8 >> 2, cc = (e8 & 3) * 8;
      cp16(Bt + l * 32 + cc, inT + ((size_t)b * Lc + lBase + l) * Ac + kb + cc);
    }
    if (kb + 32 < Ac)
      __builtin_prefetch(&W[((size_t)oBase + (tid >> 1)) * Ac + kb + 32 + (tid & 1) * 16], 0, 3);
    async_wait_all();
    __syncthreads();
    v16h a0 = load_a_frag(At, 32, 32 * wr, 0);
    v16h a1 = load_a_frag(At, 32, 32 * wr + 16, 0);
#pragma unroll
    for (int j = 0; j < 4; ++j) {
      v16h bf = load_b_frag(Bt, 32, 64 * wc + 16 * j, 0);
      acc[0][j] = wmma_f16(a0, bf, acc[0][j]);
      acc[1][j] = wmma_f16(a1, bf, acc[1][j]);
    }
  }
#pragma unroll
  for (int i = 0; i < 2; ++i)
#pragma unroll
    for (int j = 0; j < 4; ++j)
#pragma unroll
      for (int r = 0; r < 8; ++r) {
        int og = oBase + 32 * wr + 16 * i + (lane >> 4) * 8 + r;
        int lg = lBase + 64 * wc + 16 * j + (lane & 15);
        float v = acc[i][j][r] + bias[og];
        if (mode == 0) {
          outT[((size_t)b * Lc + lg) * Ac + og] = (_Float16)fmaxf(v, 0.f);
        } else {
          outF[((size_t)b * Ac + og) * Lc + lg] = v;
        }
      }
}

extern "C" void kernel_launch(void* const* d_in, const int* in_sizes, int n_in,
                              void* d_out, int out_size, void* d_ws, size_t ws_size,
                              hipStream_t stream) {
  const float* x  = (const float*)d_in[0];
  const float* Wq = (const float*)d_in[1];
  const float* bq = (const float*)d_in[2];
  const float* Wk = (const float*)d_in[3];
  const float* bk = (const float*)d_in[4];
  const float* Wv = (const float*)d_in[5];
  const float* bv = (const float*)d_in[6];
  const float* W1 = (const float*)d_in[7];
  const float* b1 = (const float*)d_in[8];
  const float* W2 = (const float*)d_in[9];
  const float* b2 = (const float*)d_in[10];

  char* ws = (char*)d_ws;
  size_t off = 0;
  _Float16* q_ws = (_Float16*)(ws + off); off += (size_t)Bc * Hc * Lc * Dc * 2;
  _Float16* k_ws = (_Float16*)(ws + off); off += (size_t)Bc * Hc * Lc * Dc * 2;
  _Float16* v_ws = (_Float16*)(ws + off); off += (size_t)Bc * Hc * Lc * Dc * 2;
  float* rmax_g  = (float*)(ws + off);    off += (size_t)Bc * Hc * Lc * 4;
  float* rsum_g  = (float*)(ws + off);    off += (size_t)Bc * Hc * Lc * 4;
  _Float16* h_ws = (_Float16*)(ws + off); off += (size_t)Bc * Lc * Ac * 2;
  _Float16* z1_ws= (_Float16*)(ws + off); off += (size_t)Bc * Lc * Ac * 2;

  dim3 blk(256);
  proj_qk_kernel<<<dim3(Lc / 128, 2 * Bc * Hc), blk, 0, stream>>>(x, Wq, bq, Wk, bk, q_ws, k_ws);
  proj_v_kernel<<<dim3(Lc / 128, Bc * Hc), blk, 0, stream>>>(x, Wv, bv, v_ws);
  attn_stats_kernel<<<dim3(Lc / 128, Bc * Hc), blk, 0, stream>>>(q_ws, k_ws, rmax_g, rsum_g);
  attn_out_kernel<<<dim3(Lc / 128, Bc * Hc), blk, 0, stream>>>(q_ws, k_ws, v_ws, rmax_g, rsum_g, h_ws);
  final_gemm_kernel<<<dim3(Lc / 128, Ac / 128, Bc), blk, 0, stream>>>(W1, b1, h_ws, z1_ws, nullptr, 0);
  final_gemm_kernel<<<dim3(Lc / 128, Ac / 128, Bc), blk, 0, stream>>>(W2, b2, z1_ws, nullptr, (float*)d_out, 1);
}
